// GIN_5325759447357
// MI455X (gfx1250) — matrix operand order repaired and checked by reading, not compiled, not run on hardware
//
#include <hip/hip_runtime.h>
#include <hip/hip_bf16.h>
#include <float.h>

// ---------------------------------------------------------------------------
// GIN / EdgeConv forward for MI455X (gfx1250).
// Heavy GEMMs: v_wmma_f32_16x16x32_f16, wave tile 16x128 (8 f32 accumulators),
// block = 256 thr = 8 waves -> block tile 128x128, grid.y = 2 (256 cols).
// BatchNorm batch statistics fused into the GEMM epilogue (shfl_xor + LDS
// cross-wave reduction + 2 atomics/column/block).  BatchNorm *apply* + ReLU
// fused into the next consumer (GEMM A-fragment load via packed f16 FMA/max,
// or the scatter-add read), eliminating the standalone streaming passes.
// ---------------------------------------------------------------------------

typedef _Float16 h16_t;
typedef _Float16 v8h  __attribute__((ext_vector_type(8)));
typedef _Float16 v16h __attribute__((ext_vector_type(16)));
typedef float    v8f  __attribute__((ext_vector_type(8)));

#define NOUT 256       // output columns of every big GEMM (= H)
#define NB   16        // NOUT / 16 column fragments (global)
#define TT   8         // T (classes)

// ---------------- small helpers -------------------------------------------

__global__ void k_fill(float* __restrict__ p, long long n, float v) {
    long long i = (long long)blockIdx.x * blockDim.x + threadIdx.x;
    if (i < n) p[i] = v;
}

__global__ void k_cvt_f32_f16(const float* __restrict__ in, h16_t* __restrict__ out,
                              long long n) {
    long long i = (long long)blockIdx.x * blockDim.x + threadIdx.x;
    if (i < n) out[i] = (h16_t)in[i];
}

__device__ __forceinline__ void atomicMaxF(float* addr, float val) {
    unsigned int* ua = (unsigned int*)addr;
    unsigned int old = __float_as_uint(*addr);
    while (__uint_as_float(old) < val) {
        unsigned int assumed = old;
        old = atomicCAS(ua, assumed, __float_as_uint(val));
        if (old == assumed) break;
    }
}

__device__ __forceinline__ v8h relu8(v8h v) {
#if defined(__has_builtin) && __has_builtin(__builtin_elementwise_max)
    const v8h z = {};
    return __builtin_elementwise_max(v, z);
#else
    v8h r;
#pragma unroll
    for (int i = 0; i < 8; ++i) r[i] = v[i] > (h16_t)0 ? v[i] : (h16_t)0;
    return r;
#endif
}

// ---------------- weight packing into WMMA B-fragment order ----------------
// Packed layout: for (kb, nb, lane, j) value = W[(kb*32 + (lane>>4)*8 + (j&7) + (j>>3)*16)*256
//                                              + nb*16 + (lane&15)]
// so each lane's 16 f16 B-fragment values are contiguous (two b128 loads).
__global__ void k_pack_b(const float* __restrict__ W, int K, h16_t* __restrict__ BP) {
    long long tid = (long long)blockIdx.x * blockDim.x + threadIdx.x;
    long long total = (long long)(K >> 5) * NB * 32 * 16;
    if (tid >= total) return;
    int j    = (int)(tid & 15);
    int lane = (int)((tid >> 4) & 31);
    int nb   = (int)((tid >> 9) & 15);
    int kb   = (int)(tid >> 13);
    int kin  = ((lane >> 4) << 3) + (j & 7) + ((j >> 3) << 4);   // K within 32-block
    int n    = (nb << 4) + (lane & 15);
    BP[tid]  = (h16_t)W[(size_t)(kb * 32 + kin) * NOUT + n];
}

// ---------------- BN stats -> per-column f16 scale/shift -------------------
__global__ void k_bnparam(const float* __restrict__ stat, const float* __restrict__ g,
                          const float* __restrict__ be, int M, h16_t* __restrict__ scsh) {
    int c = threadIdx.x;                       // 256 threads, 1 block
    float inv_m = 1.f / (float)M;
    float mean  = stat[c] * inv_m;
    float var   = stat[NOUT + c] * inv_m - mean * mean;   // biased var (jnp.var)
    float sc = g[c] * rsqrtf(var + 1e-5f);
    float sh = be[c] - mean * sc;
    scsh[c]        = (h16_t)sc;
    scsh[NOUT + c] = (h16_t)sh;
}

// ---------------- shared GEMM epilogue -------------------------------------
// Writes Y (+bias, f16), accumulates per-column sum / sumsq for BN into stat.
__device__ __forceinline__
void gemm_epilogue(v8f* acc, int M, int m0, int nbase, int wave, int lane, int khalf,
                   const float* __restrict__ bias, h16_t* __restrict__ Y,
                   float* __restrict__ stat,
                   float (*red_s)[128], float (*red_q)[128]) {
    const int rbase = m0 + khalf * 8;
#pragma unroll
    for (int j = 0; j < 8; ++j) {
        const int col = nbase + j * 16 + (lane & 15);
        const float bv = bias[col];
        float s = 0.f, q = 0.f;
#pragma unroll
        for (int r = 0; r < 8; ++r) {
            int row = rbase + r;
            float y = acc[j][r] + bv;
            if (row < M) {
                Y[(size_t)row * NOUT + col] = (h16_t)y;
                s += y; q += y * y;
            }
        }
        // combine the two row-halves (lane ^ 16 holds same column)
        s += __shfl_xor(s, 16, 32);
        q += __shfl_xor(q, 16, 32);
        if (lane < 16) {
            red_s[wave][j * 16 + lane] = s;
            red_q[wave][j * 16 + lane] = q;
        }
    }
    __syncthreads();
    const int tid = wave * 32 + lane;
    if (tid < 128) {
        float s = 0.f;
#pragma unroll
        for (int w = 0; w < 8; ++w) s += red_s[w][tid];
        atomicAdd(&stat[nbase + tid], s);
    } else {
        const int t = tid - 128;
        float q = 0.f;
#pragma unroll
        for (int w = 0; w < 8; ++w) q += red_q[w][t];
        atomicAdd(&stat[NOUT + nbase + t], q);
    }
}

// ---------------- generic WMMA GEMM ---------------------------------------
// Y[M,256](f16) = op(A[M,K]) @ Bp + bias ; accumulates BN stats of Y.
// FUSE_BN: op(a) = relu(a * sc[k] + sh[k])  (previous layer's BN applied on
// the fly to the raw pre-BN activations; packed f16 FMA/max, co-executes with
// the WMMA pipe).
template <bool FUSE_BN>
__global__ __launch_bounds__(256)
void k_gemm(const h16_t* __restrict__ A, int M, int K,
            const h16_t* __restrict__ BP, const float* __restrict__ bias,
            h16_t* __restrict__ Y, float* __restrict__ stat,
            const h16_t* __restrict__ scsh) {
    __shared__ float red_s[8][128];
    __shared__ float red_q[8][128];
    const int wave  = threadIdx.x >> 5;
    const int lane  = threadIdx.x & 31;
    const int m0    = blockIdx.x * 128 + wave * 16;
    const int khalf = lane >> 4;
    const int nbase = blockIdx.y * 128;         // this block's 8 column groups
    int ma = m0 + (lane & 15);
    if (ma >= M) ma = M - 1;                    // clamp; stores/stats guarded
    const h16_t* arow = A + (size_t)ma * K;
    const int KB = K >> 5;
    v8f z = {0.f, 0.f, 0.f, 0.f, 0.f, 0.f, 0.f, 0.f};
    v8f acc[8];
#pragma unroll
    for (int j = 0; j < 8; ++j) acc[j] = z;
    for (int kb = 0; kb < KB; ++kb) {
        const int koff = kb * 32 + khalf * 8;
        v8h alo = *(const v8h*)(arow + koff);
        v8h ahi = *(const v8h*)(arow + koff + 16);
        if (FUSE_BN) {
            v8h sl = *(const v8h*)(scsh + koff);
            v8h sh = *(const v8h*)(scsh + koff + 16);
            v8h tl = *(const v8h*)(scsh + NOUT + koff);
            v8h th = *(const v8h*)(scsh + NOUT + koff + 16);
            alo = relu8(alo * sl + tl);
            ahi = relu8(ahi * sh + th);
        }
        v16h a = __builtin_shufflevector(alo, ahi,
                 0,1,2,3,4,5,6,7,8,9,10,11,12,13,14,15);
        const h16_t* bbase = BP + (((size_t)kb * NB + (nbase >> 4)) * 32 + lane) * 16;
#pragma unroll
        for (int j = 0; j < 8; ++j) {
            const h16_t* bp = bbase + (size_t)j * 512;   // next 16-col fragment
            v8h blo = *(const v8h*)(bp);
            v8h bhi = *(const v8h*)(bp + 8);
            v16h b = __builtin_shufflevector(blo, bhi,
                     0,1,2,3,4,5,6,7,8,9,10,11,12,13,14,15);
            acc[j] = __builtin_amdgcn_wmma_f32_16x16x32_f16(false, a, false, b,
                                                            (short)0, acc[j],
                                                            false, false);
        }
    }
    gemm_epilogue(acc, M, m0, nbase, wave, lane, khalf, bias, Y, stat, red_s, red_q);
}

// ---------------- EdgeConv fused-gather GEMM -------------------------------
// A row (edge e) = [ h[dst[e]] , h[src[e]] - h[dst[e]] ]   (K = 512)
// Gathers hit L2 (h16 is 25.6 MB << 192 MB L2).
__global__ __launch_bounds__(256)
void k_gemm_edge(const h16_t* __restrict__ Hn, const int* __restrict__ src,
                 const int* __restrict__ dst, int E,
                 const h16_t* __restrict__ BP, const float* __restrict__ bias,
                 h16_t* __restrict__ Y, float* __restrict__ stat) {
    __shared__ float red_s[8][128];
    __shared__ float red_q[8][128];
    const int wave  = threadIdx.x >> 5;
    const int lane  = threadIdx.x & 31;
    const int m0    = blockIdx.x * 128 + wave * 16;
    const int khalf = lane >> 4;
    const int nbase = blockIdx.y * 128;
    int ed = m0 + (lane & 15);
    if (ed >= E) ed = E - 1;
    const h16_t* hd = Hn + (size_t)dst[ed] * NOUT;
    const h16_t* hs = Hn + (size_t)src[ed] * NOUT;
    v8f z = {0.f, 0.f, 0.f, 0.f, 0.f, 0.f, 0.f, 0.f};
    v8f acc[8];
#pragma unroll
    for (int j = 0; j < 8; ++j) acc[j] = z;
    for (int kb = 0; kb < 16; ++kb) {          // K = 512
        v8h alo, ahi;
        if (kb < 8) {                          // x_i half
            const h16_t* p = hd + kb * 32 + khalf * 8;
            alo = *(const v8h*)(p);
            ahi = *(const v8h*)(p + 16);
        } else {                               // (x_j - x_i) half
            int off = (kb - 8) * 32 + khalf * 8;
            v8h dlo = *(const v8h*)(hd + off);
            v8h dhi = *(const v8h*)(hd + off + 16);
            v8h slo = *(const v8h*)(hs + off);
            v8h shi = *(const v8h*)(hs + off + 16);
            alo = slo - dlo;                   // packed f16 subtract
            ahi = shi - dhi;
        }
        v16h a = __builtin_shufflevector(alo, ahi,
                 0,1,2,3,4,5,6,7,8,9,10,11,12,13,14,15);
        const h16_t* bbase = BP + (((size_t)kb * NB + (nbase >> 4)) * 32 + lane) * 16;
#pragma unroll
        for (int j = 0; j < 8; ++j) {
            const h16_t* bp = bbase + (size_t)j * 512;
            v8h blo = *(const v8h*)(bp);
            v8h bhi = *(const v8h*)(bp + 8);
            v16h b = __builtin_shufflevector(blo, bhi,
                     0,1,2,3,4,5,6,7,8,9,10,11,12,13,14,15);
            acc[j] = __builtin_amdgcn_wmma_f32_16x16x32_f16(false, a, false, b,
                                                            (short)0, acc[j],
                                                            false, false);
        }
    }
    gemm_epilogue(acc, E, m0, nbase, wave, lane, khalf, bias, Y, stat, red_s, red_q);
}

// ---------------- BatchNorm apply + ReLU (only where h16 must materialize) -
__global__ void k_bn_relu(const h16_t* __restrict__ in, h16_t* __restrict__ out,
                          long long total, int M, const float* __restrict__ stat,
                          const float* __restrict__ g, const float* __restrict__ be) {
    long long i = (long long)blockIdx.x * blockDim.x + threadIdx.x;
    if (i >= total) return;
    int col = (int)(i & (NOUT - 1));
    float inv_m = 1.f / (float)M;
    float mean  = stat[col] * inv_m;
    float var   = stat[NOUT + col] * inv_m - mean * mean;
    float sc = g[col] * rsqrtf(var + 1e-5f);
    float sh = be[col] - mean * sc;
    float v  = (float)in[i] * sc + sh;
    out[i]   = (h16_t)(v > 0.f ? v : 0.f);
}

// ---------------- scatter-mean with fused BN+ReLU on the message read ------
__global__ void k_scatter_add_bn(const h16_t* __restrict__ Msg, const int* __restrict__ dst,
                                 long long total, float* __restrict__ S,
                                 const float* __restrict__ stat, const float* __restrict__ g,
                                 const float* __restrict__ be, int M) {
    long long i = (long long)blockIdx.x * blockDim.x + threadIdx.x;
    if (i >= total) return;
    int e   = (int)(i >> 8);
    int col = (int)(i & (NOUT - 1));
    float inv_m = 1.f / (float)M;
    float mean  = stat[col] * inv_m;
    float var   = stat[NOUT + col] * inv_m - mean * mean;
    float sc = g[col] * rsqrtf(var + 1e-5f);
    float sh = be[col] - mean * sc;
    float v  = (float)Msg[i] * sc + sh;
    v = v > 0.f ? v : 0.f;
    atomicAdd(&S[(size_t)dst[e] * NOUT + col], v);
}

__global__ void k_count(const int* __restrict__ dst, int E, float* __restrict__ cnt) {
    int e = blockIdx.x * blockDim.x + threadIdx.x;
    if (e < E) atomicAdd(&cnt[dst[e]], 1.f);
}

__global__ void k_divide(const float* __restrict__ S, const float* __restrict__ cnt,
                         long long total, h16_t* __restrict__ Hn) {
    long long i = (long long)blockIdx.x * blockDim.x + threadIdx.x;
    if (i >= total) return;
    float c = cnt[i >> 8];
    Hn[i] = (h16_t)(S[i] / (c > 1.f ? c : 1.f));
}

// ---------------- per-layer head: z = h@lw + lb;  Z += z;  segment max -----
__global__ void k_zlayer(const h16_t* __restrict__ Hn, const float* __restrict__ lw,
                         const float* __restrict__ lb, const int* __restrict__ batch,
                         int N, float* __restrict__ Zacc, float* __restrict__ maxbuf) {
    int i = blockIdx.x * blockDim.x + threadIdx.x;
    if (i >= N) return;
    const h16_t* hr = Hn + (size_t)i * NOUT;
    float acc[TT];
#pragma unroll
    for (int t = 0; t < TT; ++t) acc[t] = 0.f;
    for (int k = 0; k < NOUT; ++k) {
        float hv = (float)hr[k];
#pragma unroll
        for (int t = 0; t < TT; ++t) acc[t] += hv * lw[k * TT + t];
    }
    int b = batch[i];
#pragma unroll
    for (int t = 0; t < TT; ++t) {
        float z = acc[t] + lb[t];
        Zacc[(size_t)i * TT + t] += z;
        atomicMaxF(&maxbuf[b * TT + t], z);
    }
}

__global__ void k_addmax(float* __restrict__ out, const float* __restrict__ maxbuf, int n) {
    int i = blockIdx.x * blockDim.x + threadIdx.x;
    if (i < n) out[i] += maxbuf[i];
}

// ---------------------------------------------------------------------------

extern "C" void kernel_launch(void* const* d_in, const int* in_sizes, int n_in,
                              void* d_out, int out_size, void* d_ws, size_t ws_size,
                              hipStream_t stream) {
    const float* x     = (const float*)d_in[0];
    const int*   ei    = (const int*)d_in[1];
    const int*   batch = (const int*)d_in[2];
    const float* fw1 = (const float*)d_in[3];  const float* fb1  = (const float*)d_in[4];
    const float* fg1 = (const float*)d_in[5];  const float* fbe1 = (const float*)d_in[6];
    const float* fw2 = (const float*)d_in[7];  const float* fb2  = (const float*)d_in[8];
    const float* fg2 = (const float*)d_in[9];  const float* fbe2 = (const float*)d_in[10];
    const float* lws[3] = {(const float*)d_in[11], (const float*)d_in[13], (const float*)d_in[15]};
    const float* lbs[3] = {(const float*)d_in[12], (const float*)d_in[14], (const float*)d_in[16]};
    const float* ew1[2] = {(const float*)d_in[17], (const float*)d_in[25]};
    const float* eb1[2] = {(const float*)d_in[18], (const float*)d_in[26]};
    const float* eg1[2] = {(const float*)d_in[19], (const float*)d_in[27]};
    const float* ee1[2] = {(const float*)d_in[20], (const float*)d_in[28]};
    const float* ew2[2] = {(const float*)d_in[21], (const float*)d_in[29]};
    const float* eb2[2] = {(const float*)d_in[22], (const float*)d_in[30]};
    const float* eg2[2] = {(const float*)d_in[23], (const float*)d_in[31]};
    const float* ee2[2] = {(const float*)d_in[24], (const float*)d_in[32]};

    const int N = in_sizes[2];
    const int E = in_sizes[1] / 2;
    const int F = in_sizes[0] / N;     // 128
    const int G = (out_size - N * TT) / TT;
    const int* src = ei;
    const int* dst = ei + E;

    // workspace carve (256B aligned)
    char* wp = (char*)d_ws;
    auto carve = [&](size_t bytes) -> void* {
        void* r = (void*)wp;
        wp += (bytes + 255) & ~(size_t)255;
        return r;
    };
    h16_t* x16   = (h16_t*)carve((size_t)N * F * sizeof(h16_t));
    h16_t* h16   = (h16_t*)carve((size_t)N * NOUT * sizeof(h16_t));
    h16_t* tA    = (h16_t*)carve((size_t)E * NOUT * sizeof(h16_t));
    h16_t* tB    = (h16_t*)carve((size_t)E * NOUT * sizeof(h16_t));
    h16_t* bpack = (h16_t*)carve((size_t)512 * NOUT * sizeof(h16_t));
    float* stat  = (float*)carve(2 * NOUT * sizeof(float));
    h16_t* scsh  = (h16_t*)carve(2 * NOUT * sizeof(h16_t));
    float* S     = (float*)carve((size_t)N * NOUT * sizeof(float));
    float* cnt   = (float*)carve((size_t)N * sizeof(float));
    float* mbuf  = (float*)carve((size_t)G * TT * sizeof(float));

    float* out = (float*)d_out;        // [G,T]
    float* Z   = out + (size_t)G * TT; // [N,T]

    const int BS = 256;
    auto blocksLL = [&](long long n) { return (unsigned)((n + BS - 1) / BS); };

    // init outputs
    k_fill<<<blocksLL((long long)G * TT + (long long)N * TT), BS, 0, stream>>>(
        out, (long long)G * TT + (long long)N * TT, 0.f);
    // x -> f16
    k_cvt_f32_f16<<<blocksLL((long long)N * F), BS, 0, stream>>>(x, x16, (long long)N * F);

    dim3 gemm_grid_N((N + 127) / 128, 2);
    dim3 gemm_grid_E((E + 127) / 128, 2);

    // ---------------- layer 0: first_h MLP ----------------
    k_pack_b<<<blocksLL((long long)F * NOUT), BS, 0, stream>>>(fw1, F, bpack);
    k_fill<<<blocksLL(2 * NOUT), BS, 0, stream>>>(stat, 2 * NOUT, 0.f);
    k_gemm<false><<<gemm_grid_N, BS, 0, stream>>>(x16, N, F, bpack, fb1, tA, stat,
                                                  (const h16_t*)nullptr);
    k_bnparam<<<1, NOUT, 0, stream>>>(stat, fg1, fbe1, N, scsh);

    k_pack_b<<<blocksLL((long long)NOUT * NOUT), BS, 0, stream>>>(fw2, NOUT, bpack);
    k_fill<<<blocksLL(2 * NOUT), BS, 0, stream>>>(stat, 2 * NOUT, 0.f);
    k_gemm<true><<<gemm_grid_N, BS, 0, stream>>>(tA, N, NOUT, bpack, fb2, tB, stat, scsh);
    k_bn_relu<<<blocksLL((long long)N * NOUT), BS, 0, stream>>>(
        tB, h16, (long long)N * NOUT, N, stat, fg2, fbe2);

    k_fill<<<blocksLL((long long)G * TT), BS, 0, stream>>>(mbuf, (long long)G * TT, -3.4e38f);
    k_zlayer<<<blocksLL(N), BS, 0, stream>>>(h16, lws[0], lbs[0], batch, N, Z, mbuf);
    k_addmax<<<blocksLL((long long)G * TT), BS, 0, stream>>>(out, mbuf, G * TT);

    // ---------------- EdgeConv layers 1 & 2 ----------------
    for (int l = 0; l < 2; ++l) {
        // GEMM1: [E,512] (fused gather) x [512,256] -> raw pre-BN tA + stats
        k_pack_b<<<blocksLL((long long)512 * NOUT), BS, 0, stream>>>(ew1[l], 512, bpack);
        k_fill<<<blocksLL(2 * NOUT), BS, 0, stream>>>(stat, 2 * NOUT, 0.f);
        k_gemm_edge<<<gemm_grid_E, BS, 0, stream>>>(h16, src, dst, E, bpack, eb1[l],
                                                    tA, stat);
        k_bnparam<<<1, NOUT, 0, stream>>>(stat, eg1[l], ee1[l], E, scsh);

        // GEMM2: [E,256] x [256,256], BN1+ReLU applied on A-load -> raw tB + stats
        k_pack_b<<<blocksLL((long long)NOUT * NOUT), BS, 0, stream>>>(ew2[l], NOUT, bpack);
        k_fill<<<blocksLL(2 * NOUT), BS, 0, stream>>>(stat, 2 * NOUT, 0.f);
        k_gemm<true><<<gemm_grid_E, BS, 0, stream>>>(tA, E, NOUT, bpack, eb2[l], tB,
                                                     stat, scsh);

        // scatter-mean over dst, BN2+ReLU fused into the message read
        k_fill<<<blocksLL((long long)N * NOUT), BS, 0, stream>>>(S, (long long)N * NOUT, 0.f);
        k_fill<<<blocksLL(N), BS, 0, stream>>>(cnt, N, 0.f);
        k_scatter_add_bn<<<blocksLL((long long)E * NOUT), BS, 0, stream>>>(
            tB, dst, (long long)E * NOUT, S, stat, eg2[l], ee2[l], E);
        k_count<<<blocksLL(E), BS, 0, stream>>>(dst, E, cnt);
        k_divide<<<blocksLL((long long)N * NOUT), BS, 0, stream>>>(
            S, cnt, (long long)N * NOUT, h16);

        // head
        k_fill<<<blocksLL((long long)G * TT), BS, 0, stream>>>(mbuf, (long long)G * TT, -3.4e38f);
        k_zlayer<<<blocksLL(N), BS, 0, stream>>>(h16, lws[l + 1], lbs[l + 1], batch, N, Z, mbuf);
        k_addmax<<<blocksLL((long long)G * TT), BS, 0, stream>>>(out, mbuf, G * TT);
    }
    (void)n_in; (void)ws_size;
}
